// ReferenceAttention_59502476918792
// MI455X (gfx1250) — matrix operand order repaired and checked by reading, not compiled, not run on hardware
//
#include <hip/hip_runtime.h>

typedef __attribute__((ext_vector_type(16))) _Float16 v16h;
typedef __attribute__((ext_vector_type(8)))  _Float16 v8h;
typedef __attribute__((ext_vector_type(8)))  float    v8f;

#define S_LEN   2048
#define HIDDEN  2048
#define NH      32
#define NKV     8
#define DH      64
#define QKV_N   3072          // NH*DH + 2*NKV*DH
#define KVW     512           // NKV*DH
#define HALF_D  32
#define ATT_SCALE 0.125f      // 64^-0.5
#define NEGINF  (-1.0e9f)

// ---------------------------------------------------------------- helpers

__device__ __forceinline__ v16h mk16(v8h lo, v8h hi) {
  v16h r;
#pragma unroll
  for (int i = 0; i < 8; ++i) { r[i] = lo[i]; r[i + 8] = hi[i]; }
  return r;
}

__device__ __forceinline__ v8f zero8() {
  v8f z;
#pragma unroll
  for (int i = 0; i < 8; ++i) z[i] = 0.0f;
  return z;
}

__device__ __forceinline__ v8f wmma16(v16h a, v16h b, v8f c) {
  // (neg_a, A, neg_b, B, c_mod, C, reuse_a, reuse_b)
  return __builtin_amdgcn_wmma_f32_16x16x32_f16(false, a, false, b, (short)0, c,
                                                false, false);
}

// ---------------------------------------------------------------- elementwise

__global__ void cvt_f32_f16_kernel(const float* __restrict__ src,
                                   _Float16* __restrict__ dst, int n) {
  int i = blockIdx.x * 256 + threadIdx.x;
  if (i < n) dst[i] = (_Float16)src[i];
}

// Pack Wq[2048x2048] | Wk[2048x512] | Wv[2048x512] -> f16 [2048 x 3072]
__global__ void pack_qkv_f16_kernel(const float* __restrict__ Wq,
                                    const float* __restrict__ Wk,
                                    const float* __restrict__ Wv,
                                    _Float16* __restrict__ dst) {
  int i = blockIdx.x * 256 + threadIdx.x;
  if (i >= HIDDEN * QKV_N) return;
  int r = i / QKV_N;
  int c = i - r * QKV_N;
  float v;
  if (c < HIDDEN)            v = Wq[(size_t)r * HIDDEN + c];
  else if (c < HIDDEN + KVW) v = Wk[(size_t)r * KVW + (c - HIDDEN)];
  else                       v = Wv[(size_t)r * KVW + (c - HIDDEN - KVW)];
  dst[i] = (_Float16)v;
}

// Apply RoPE to q/k columns of the fused QKV result and narrow to f16.
__global__ void rope_cvt_kernel(const float* __restrict__ qkv,
                                const float* __restrict__ cosb,
                                const float* __restrict__ sinb,
                                _Float16* __restrict__ q16,
                                _Float16* __restrict__ k16,
                                _Float16* __restrict__ v16o) {
  int i = blockIdx.x * 256 + threadIdx.x;
  if (i >= S_LEN * QKV_N) return;
  int s = i / QKV_N;
  int c = i - s * QKV_N;
  float val = qkv[i];
  if (c < HIDDEN) {                       // q, head = c/64
    int d = c & (DH - 1);
    float out;
    if (d < HALF_D)
      out = val * cosb[s * HALF_D + d] - qkv[i + HALF_D] * sinb[s * HALF_D + d];
    else
      out = val * cosb[s * HALF_D + d - HALF_D] +
            qkv[i - HALF_D] * sinb[s * HALF_D + d - HALF_D];
    q16[(size_t)s * HIDDEN + c] = (_Float16)out;
  } else if (c < HIDDEN + KVW) {          // k
    int cc = c - HIDDEN;
    int d = cc & (DH - 1);
    float out;
    if (d < HALF_D)
      out = val * cosb[s * HALF_D + d] - qkv[i + HALF_D] * sinb[s * HALF_D + d];
    else
      out = val * cosb[s * HALF_D + d - HALF_D] +
            qkv[i - HALF_D] * sinb[s * HALF_D + d - HALF_D];
    k16[(size_t)s * KVW + cc] = (_Float16)out;
  } else {                                // v (no rope)
    v16o[(size_t)s * KVW + (c - HIDDEN - KVW)] = (_Float16)val;
  }
}

// ---------------------------------------------------------------- generic WMMA GEMM
// C[MxN] (f32) = A[MxK] (f16, row-major) * B[KxN] (f16, row-major)
// block = 256 threads (8 waves), tile 128x128, BK=32.
// wave grid 4x2: each wave computes a 32x64 sub-tile = 2x4 WMMA accumulators.

__global__ void wmma_gemm_f16_kernel(const _Float16* __restrict__ A,
                                     const _Float16* __restrict__ B,
                                     float* __restrict__ C,
                                     int M, int N, int K) {
  __shared__ _Float16 As[128][32];
  __shared__ _Float16 Bs[32][128];

  const int tid  = threadIdx.x;
  const int wave = tid >> 5;
  const int lane = tid & 31;
  const int wm   = wave >> 1;       // 0..3 -> row offset wm*32
  const int wn   = wave & 1;        // 0..1 -> col offset wn*64
  const int mlo  = lane & 15;
  const int hi   = lane >> 4;

  const int rowBase = blockIdx.y * 128;
  const int colBase = blockIdx.x * 128;

  v8f acc[2][4];
#pragma unroll
  for (int a = 0; a < 2; ++a)
#pragma unroll
    for (int b = 0; b < 4; ++b) acc[a][b] = zero8();

  for (int k0 = 0; k0 < K; k0 += 32) {
    {   // stage A tile 128x32 (coalesced, 16 halfs / thread)
      int r = tid >> 1, c = (tid & 1) * 16;
      const _Float16* src = A + (size_t)(rowBase + r) * K + k0 + c;
      *(v8h*)&As[r][c]     = *(const v8h*)src;
      *(v8h*)&As[r][c + 8] = *(const v8h*)(src + 8);
    }
    {   // stage B tile 32x128 (coalesced)
      int r = tid >> 3, c = (tid & 7) * 16;
      const _Float16* src = B + (size_t)(k0 + r) * N + colBase + c;
      *(v8h*)&Bs[r][c]     = *(const v8h*)src;
      *(v8h*)&Bs[r][c + 8] = *(const v8h*)(src + 8);
    }
    __syncthreads();

#pragma unroll
    for (int tm = 0; tm < 2; ++tm) {
      const _Float16* ap = &As[wm * 32 + tm * 16 + mlo][hi * 8];
      v16h af = mk16(*(const v8h*)ap, *(const v8h*)(ap + 16));
#pragma unroll
      for (int tn = 0; tn < 4; ++tn) {
        v16h bf;
#pragma unroll
        for (int s = 0; s < 16; ++s)
          bf[s] = Bs[hi * 8 + s + (s >= 8 ? 8 : 0)][wn * 64 + tn * 16 + mlo];
        acc[tm][tn] = wmma16(af, bf, acc[tm][tn]);
      }
    }
    __syncthreads();
  }

#pragma unroll
  for (int tm = 0; tm < 2; ++tm)
#pragma unroll
    for (int tn = 0; tn < 4; ++tn)
#pragma unroll
      for (int i = 0; i < 8; ++i) {
        int r = rowBase + wm * 32 + tm * 16 + hi * 8 + i;
        int c = colBase + wn * 64 + tn * 16 + mlo;
        C[(size_t)r * N + c] = acc[tm][tn][i];
      }
}

// ---------------------------------------------------------------- attention scores
// grid = (kblk 16, qblk 16, head 32), block = 256 (8 waves).
// Each block: 128q x 128k score tile for one head, via WMMA over D=64 (2 k-steps).
// Fully-masked blocks write the FINAL probability value (exactly 0) so no later
// pass ever touches the masked half again.

__global__ void attn_scores_kernel(const _Float16* __restrict__ q16,
                                   const _Float16* __restrict__ k16,
                                   float* __restrict__ Pout) {
  const int kbase = blockIdx.x * 128;
  const int qb0   = blockIdx.y * 128;
  const int h     = blockIdx.z;

  float* out = Pout + ((size_t)h * S_LEN + qb0) * S_LEN + kbase;

  if (kbase > qb0 + 127) {  // fully masked: final prob is exactly 0
    for (int i = threadIdx.x; i < 128 * 128; i += 256) {
      int r = i >> 7, c = i & 127;
      out[(size_t)r * S_LEN + c] = 0.0f;
    }
    return;
  }

  __shared__ _Float16 kT[64][128];   // K transposed: [d][key]
  const int hkv = h >> 2;            // GQA: 4 q-heads per kv-head
  for (int i = threadIdx.x; i < 128 * 64; i += 256) {
    int krow = i >> 6, d = i & 63;
    kT[d][krow] = k16[(size_t)(kbase + krow) * KVW + hkv * DH + d];
  }
  __syncthreads();

  const int wave = threadIdx.x >> 5;
  const int lane = threadIdx.x & 31;
  const int mlo  = lane & 15;
  const int hi   = lane >> 4;
  const int qbase = qb0 + wave * 16;

  v8f acc[8];
#pragma unroll
  for (int t = 0; t < 8; ++t) acc[t] = zero8();

#pragma unroll
  for (int kk = 0; kk < 64; kk += 32) {
    const _Float16* ap = q16 + (size_t)(qbase + mlo) * HIDDEN + h * DH + kk + hi * 8;
    v16h af = mk16(*(const v8h*)ap, *(const v8h*)(ap + 16));
#pragma unroll
    for (int t = 0; t < 8; ++t) {
      v16h bf;
#pragma unroll
      for (int s = 0; s < 16; ++s)
        bf[s] = kT[kk + hi * 8 + s + (s >= 8 ? 8 : 0)][t * 16 + mlo];
      acc[t] = wmma16(af, bf, acc[t]);
    }
  }

#pragma unroll
  for (int t = 0; t < 8; ++t)
#pragma unroll
    for (int i = 0; i < 8; ++i) {
      int qr = qbase + hi * 8 + i;
      int kc = kbase + t * 16 + mlo;
      float sv = acc[t][i] * ATT_SCALE + (qr >= kc ? 0.0f : NEGINF);
      Pout[((size_t)h * S_LEN + qr) * S_LEN + kc] = sv;
    }
}

// ---------------------------------------------------------------- row stats
// one 256-thread block per (head, query) row; reads only the causal prefix
// (cols 0..q), produces per-row (max, 1/denom) including the sink logit.

__global__ void row_stats_kernel(const float* __restrict__ Sc,
                                 const float* __restrict__ sinks,
                                 float* __restrict__ stats /* [NH*S][2] */) {
  const int row = blockIdx.x;            // h*S + q
  const int h   = row >> 11;
  const int q   = row & (S_LEN - 1);
  const float* p = Sc + (size_t)row * S_LEN;
  const int n = q + 1;                   // causal prefix length
  __shared__ float red[256];
  const int tid = threadIdx.x;

  float snk = sinks[h];
  float mx = snk;
  for (int i = tid; i < n; i += 256) mx = fmaxf(mx, p[i]);
  red[tid] = mx;
  __syncthreads();
  for (int s = 128; s > 0; s >>= 1) {
    if (tid < s) red[tid] = fmaxf(red[tid], red[tid + s]);
    __syncthreads();
  }
  mx = red[0];
  __syncthreads();

  float sum = 0.0f;
  for (int i = tid; i < n; i += 256) sum += __expf(p[i] - mx);
  red[tid] = sum;
  __syncthreads();
  for (int s = 128; s > 0; s >>= 1) {
    if (tid < s) red[tid] += red[tid + s];
    __syncthreads();
  }
  if (tid == 0) {
    float denom = red[0] + __expf(snk - mx);
    stats[(size_t)row * 2]     = mx;
    stats[(size_t)row * 2 + 1] = 1.0f / denom;
  }
}

// ---------------------------------------------------------------- fused softmax + P @ V
// grid = (qblk 16, head 32), block 256 (8 waves). Wave = one 16q x 64d tile.
// Reads raw scores in the WMMA A-fragment layout, applies exp((s-m))*inv in
// registers, writes the final probabilities back in place (causal band only),
// converts to f16 and accumulates P@V with WMMA. Causal loop bound skips the
// zero half entirely; a small cleanup zero-fills the in-band masked sliver.

__global__ void attn_pv_kernel(float* __restrict__ P,
                               const float* __restrict__ stats,
                               const _Float16* __restrict__ v16b,
                               float* __restrict__ Oout) {
  const int h    = blockIdx.y;
  const int wave = threadIdx.x >> 5;
  const int lane = threadIdx.x & 31;
  const int qb0  = blockIdx.x * 128;
  const int qbase = qb0 + wave * 16;
  const int hkv  = h >> 2;
  const int mlo  = lane & 15;
  const int hi   = lane >> 4;

  const int   qrow = qbase + mlo;                       // this lane's A-row
  const float m    = stats[((size_t)h * S_LEN + qrow) * 2];
  const float inv  = stats[((size_t)h * S_LEN + qrow) * 2 + 1];

  v8f acc[4];
#pragma unroll
  for (int t = 0; t < 4; ++t) acc[t] = zero8();

  float* prow = P + ((size_t)h * S_LEN + qrow) * S_LEN;
  const int kend = qbase + 16;   // max k needed by any row of this wave's tile

  int k0;
  for (k0 = 0; k0 < kend; k0 += 32) {
    float* pp = prow + k0 + hi * 8;
    float4 f0 = *(const float4*)(pp);
    float4 f1 = *(const float4*)(pp + 4);
    float4 f2 = *(const float4*)(pp + 16);
    float4 f3 = *(const float4*)(pp + 20);

    // softmax in registers (NEG-masked entries underflow to exactly 0)
    f0.x = __expf(f0.x - m) * inv; f0.y = __expf(f0.y - m) * inv;
    f0.z = __expf(f0.z - m) * inv; f0.w = __expf(f0.w - m) * inv;
    f1.x = __expf(f1.x - m) * inv; f1.y = __expf(f1.y - m) * inv;
    f1.z = __expf(f1.z - m) * inv; f1.w = __expf(f1.w - m) * inv;
    f2.x = __expf(f2.x - m) * inv; f2.y = __expf(f2.y - m) * inv;
    f2.z = __expf(f2.z - m) * inv; f2.w = __expf(f2.w - m) * inv;
    f3.x = __expf(f3.x - m) * inv; f3.y = __expf(f3.y - m) * inv;
    f3.z = __expf(f3.z - m) * inv; f3.w = __expf(f3.w - m) * inv;

    // write final probabilities back in place
    *(float4*)(pp)      = f0;
    *(float4*)(pp + 4)  = f1;
    *(float4*)(pp + 16) = f2;
    *(float4*)(pp + 20) = f3;

    v16h af;
    af[0]  = (_Float16)f0.x; af[1]  = (_Float16)f0.y;
    af[2]  = (_Float16)f0.z; af[3]  = (_Float16)f0.w;
    af[4]  = (_Float16)f1.x; af[5]  = (_Float16)f1.y;
    af[6]  = (_Float16)f1.z; af[7]  = (_Float16)f1.w;
    af[8]  = (_Float16)f2.x; af[9]  = (_Float16)f2.y;
    af[10] = (_Float16)f2.z; af[11] = (_Float16)f2.w;
    af[12] = (_Float16)f3.x; af[13] = (_Float16)f3.y;
    af[14] = (_Float16)f3.z; af[15] = (_Float16)f3.w;

    const _Float16* vrow = v16b + (size_t)k0 * KVW + hkv * DH + mlo;
#pragma unroll
    for (int t = 0; t < 4; ++t) {
      v16h bf;
#pragma unroll
      for (int s = 0; s < 16; ++s)
        bf[s] = vrow[(size_t)(hi * 8 + s + (s >= 8 ? 8 : 0)) * KVW + t * 16];
      acc[t] = wmma16(af, bf, acc[t]);
    }
  }

  // store O tile
#pragma unroll
  for (int t = 0; t < 4; ++t)
#pragma unroll
    for (int i = 0; i < 8; ++i) {
      int qr = qbase + hi * 8 + i;
      Oout[(size_t)qr * HIDDEN + h * DH + t * 16 + mlo] = acc[t][i];
    }

  // zero-fill the in-band masked sliver: cols [K1, qb0+128) for this wave's rows
  const int K1    = k0;                 // roundup32(kend)
  const int width = qb0 + 128 - K1;
  for (int i = lane; i < 16 * width; i += 32) {
    int r = i / width, c = i - r * width;
    P[((size_t)h * S_LEN + qbase + r) * S_LEN + K1 + c] = 0.0f;
  }
}

// ---------------------------------------------------------------- launch

extern "C" void kernel_launch(void* const* d_in, const int* in_sizes, int n_in,
                              void* d_out, int out_size, void* d_ws, size_t ws_size,
                              hipStream_t stream) {
  const float* x     = (const float*)d_in[0];   // [1,2048,2048]
  const float* cosb  = (const float*)d_in[1];   // [1,2048,32]
  const float* sinb  = (const float*)d_in[2];   // [1,2048,32]
  // d_in[3] = attention_mask (causal mask recomputed analytically)
  const float* Wq    = (const float*)d_in[4];
  const float* Wk    = (const float*)d_in[5];
  const float* Wv    = (const float*)d_in[6];
  const float* Wo    = (const float*)d_in[7];
  const float* sinks = (const float*)d_in[8];

  float* out_attn = (float*)d_out;                                   // [2048,2048]
  float* out_w    = (float*)d_out + (size_t)S_LEN * HIDDEN;          // [32,2048,2048]

  // workspace carve-up
  char* ws = (char*)d_ws;
  _Float16* x16     = (_Float16*)ws;                      ws += (size_t)S_LEN * HIDDEN * 2;
  _Float16* wqkv16  = (_Float16*)ws;                      ws += (size_t)HIDDEN * QKV_N * 2;
  _Float16* wo16    = (_Float16*)ws;                      ws += (size_t)HIDDEN * HIDDEN * 2;
  float*    qkv     = (float*)ws;                         ws += (size_t)S_LEN * QKV_N * 4;
  _Float16* q16     = (_Float16*)ws;                      ws += (size_t)S_LEN * HIDDEN * 2;
  _Float16* k16     = (_Float16*)ws;                      ws += (size_t)S_LEN * KVW * 2;
  _Float16* v16b    = (_Float16*)ws;                      ws += (size_t)S_LEN * KVW * 2;
  float*    attnout = (float*)ws;                         ws += (size_t)S_LEN * HIDDEN * 4;
  _Float16* attnout16 = (_Float16*)ws;                    ws += (size_t)S_LEN * HIDDEN * 2;
  float*    stats   = (float*)ws;   // [NH*S][2] = 512 KB

  const int nx   = S_LEN * HIDDEN;     // 4 Mi elems
  const int nqkv = HIDDEN * QKV_N;     // 6 Mi elems

  // 1) precision conversion / weight packing
  cvt_f32_f16_kernel<<<(nx + 255) / 256, 256, 0, stream>>>(x, x16, nx);
  pack_qkv_f16_kernel<<<(nqkv + 255) / 256, 256, 0, stream>>>(Wq, Wk, Wv, wqkv16);
  cvt_f32_f16_kernel<<<(nx + 255) / 256, 256, 0, stream>>>(Wo, wo16, nx);

  // 2) fused QKV projection: [2048x2048] x [2048x3072]
  wmma_gemm_f16_kernel<<<dim3(QKV_N / 128, S_LEN / 128), 256, 0, stream>>>(
      x16, wqkv16, qkv, S_LEN, QKV_N, HIDDEN);

  // 3) RoPE + narrow q/k/v to f16
  {
    const int n = S_LEN * QKV_N;
    rope_cvt_kernel<<<(n + 255) / 256, 256, 0, stream>>>(qkv, cosb, sinb,
                                                         q16, k16, v16b);
  }

  // 4) raw scores (masked half written as final 0-probabilities)
  attn_scores_kernel<<<dim3(16, 16, NH), 256, 0, stream>>>(q16, k16, out_w);

  // 5) per-row softmax stats over the causal prefix + sink
  row_stats_kernel<<<NH * S_LEN, 256, 0, stream>>>(out_w, sinks, stats);

  // 6) fused softmax + O = P V (writes final probabilities in place)
  attn_pv_kernel<<<dim3(16, NH), 256, 0, stream>>>(out_w, stats, v16b, attnout);

  // 7) output projection: [2048x2048] x [2048x2048] -> d_out
  cvt_f32_f16_kernel<<<(nx + 255) / 256, 256, 0, stream>>>(attnout, attnout16, nx);
  wmma_gemm_f16_kernel<<<dim3(HIDDEN / 128, S_LEN / 128), 256, 0, stream>>>(
      attnout16, wo16, out_attn, S_LEN, HIDDEN, HIDDEN);
}
